// KEP_GAT_PNA_47880295416184
// MI455X (gfx1250) — compile-verified, matrix operand order checked by reading, not run on hardware
//
#include <hip/hip_runtime.h>
#include <hip/hip_bf16.h>
#include <math.h>

// ---------------- problem constants (from reference) ----------------
#define NN      50000
#define EE      1600000
#define ESL     1650000          // EE + NN self loops
#define INV_AVG_DEG_LOG 0.28599869177739284f  // 1/ln(33)

typedef __attribute__((ext_vector_type(16))) _Float16 v16h;
typedef __attribute__((ext_vector_type(8)))  float    v8f;

// ---------------- helpers: order-preserving float<->uint for atomic min/max ----------------
__device__ __forceinline__ unsigned fenc(float f) {
    unsigned u = __float_as_uint(f);
    return (u & 0x80000000u) ? ~u : (u | 0x80000000u);
}
__device__ __forceinline__ float fdec(unsigned u) {
    unsigned v = (u & 0x80000000u) ? (u & 0x7fffffffu) : ~u;
    return __uint_as_float(v);
}
__device__ __forceinline__ float lrelu(float x) { return x > 0.f ? x : 0.2f * x; }
__device__ __forceinline__ float relu(float x)  { return x > 0.f ? x : 0.f; }

// ---------------- init all accumulators ----------------
__global__ void k_init(float* pna_sum, float* pna_sq, unsigned* pna_min, unsigned* pna_max,
                       float* cnt, float* easum,
                       unsigned* max1, float* sum1, float* out1,
                       unsigned* max2, float* sum2, float* out2,
                       unsigned* smax, float* ssum) {
    int i = blockIdx.x * blockDim.x + threadIdx.x;
    if (i < NN * 5)  { pna_sum[i] = 0.f; pna_sq[i] = 0.f; pna_min[i] = 0xFFFFFFFFu; pna_max[i] = 0u; }
    if (i < NN)      { cnt[i] = 0.f; easum[i] = 0.f; }
    if (i < NN * 4)  { max1[i] = 0u; sum1[i] = 0.f; max2[i] = 0u; sum2[i] = 0.f; }
    if (i < NN * 16) { out1[i] = 0.f; }
    if (i < NN * 32) { out2[i] = 0.f; }
    if (i < NN * 2)  { smax[i] = 0u; ssum[i] = 0.f; }
}

// ---------------- PNA edge phase ----------------
__global__ void k_pna_edge(const float* __restrict__ x, const int* __restrict__ src,
                           const int* __restrict__ dst, const float* __restrict__ ew,
                           const float* __restrict__ W_ee, const float* __restrict__ b_ee,
                           const float* __restrict__ W_pre, const float* __restrict__ b_pre,
                           float* pna_sum, float* pna_sq, unsigned* pna_min, unsigned* pna_max,
                           float* cnt, float* easum) {
    int e = blockIdx.x * blockDim.x + threadIdx.x;
    if (e >= EE) return;
    int s = src[e], d = dst[e];
    float ea = ew[e];
    float h[15];
    #pragma unroll
    for (int j = 0; j < 5; ++j) h[j]      = x[(size_t)d * 5 + j];
    #pragma unroll
    for (int j = 0; j < 5; ++j) h[5 + j]  = x[(size_t)s * 5 + j];
    #pragma unroll
    for (int j = 0; j < 5; ++j) h[10 + j] = ea * W_ee[j] + b_ee[j];
    #pragma unroll
    for (int j = 0; j < 5; ++j) {
        float m = b_pre[j];
        #pragma unroll
        for (int k = 0; k < 15; ++k) m += h[k] * W_pre[k * 5 + j];
        atomicAdd(&pna_sum[(size_t)d * 5 + j], m);
        atomicAdd(&pna_sq [(size_t)d * 5 + j], m * m);
        atomicMin(&pna_min[(size_t)d * 5 + j], fenc(m));
        atomicMax(&pna_max[(size_t)d * 5 + j], fenc(m));
    }
    atomicAdd(&cnt[d], 1.f);
    atomicAdd(&easum[d], ea);
}

// ---------------- PNA node phase: aggregate -> 65 feats -> W_post -> W_pna ----------------
__global__ void k_pna_node(const float* __restrict__ x,
                           const float* pna_sum, const float* pna_sq,
                           const unsigned* pna_min, const unsigned* pna_max,
                           const float* cnt, const float* easum,
                           const float* __restrict__ W_post, const float* __restrict__ b_post,
                           const float* __restrict__ W_pna, const float* __restrict__ b_pna,
                           float* __restrict__ nf8, float* __restrict__ eamean) {
    int i = blockIdx.x * blockDim.x + threadIdx.x;
    if (i >= NN) return;
    float cn = cnt[i];
    float cc = fmaxf(cn, 1.f);
    bool has = cn > 0.f;
    float agg[20];
    #pragma unroll
    for (int j = 0; j < 5; ++j) {
        float me = pna_sum[(size_t)i * 5 + j] / cc;
        float ms = pna_sq [(size_t)i * 5 + j] / cc;
        float var = fmaxf(ms - me * me, 0.f);
        agg[j]      = me;
        agg[5 + j]  = has ? fdec(pna_min[(size_t)i * 5 + j]) : 0.f;
        agg[10 + j] = has ? fdec(pna_max[(size_t)i * 5 + j]) : 0.f;
        agg[15 + j] = sqrtf(var + 1e-5f);
    }
    float amp = logf(cc + 1.f) * INV_AVG_DEG_LOG;
    float o[65];
    #pragma unroll
    for (int j = 0; j < 5; ++j)  o[j] = x[(size_t)i * 5 + j];
    #pragma unroll
    for (int k = 0; k < 20; ++k) { o[5 + k] = agg[k]; o[25 + k] = agg[k] * amp; o[45 + k] = agg[k] / amp; }
    float t8[8];
    #pragma unroll
    for (int j = 0; j < 8; ++j) {
        float v = b_post[j];
        for (int k = 0; k < 65; ++k) v += o[k] * W_post[k * 8 + j];
        t8[j] = v;
    }
    #pragma unroll
    for (int j = 0; j < 8; ++j) {
        float v = b_pna[j];
        #pragma unroll
        for (int k = 0; k < 8; ++k) v += t8[k] * W_pna[k * 8 + j];
        nf8[(size_t)i * 8 + j] = v;
    }
    eamean[i] = easum[i] / cc;
}

// ---------------- dual small linear (xl, xr) ----------------
template <int IN, int OUT>
__global__ void k_dual_linear(const float* __restrict__ in,
                              const float* __restrict__ Wl, const float* __restrict__ bl,
                              const float* __restrict__ Wr, const float* __restrict__ br,
                              float* __restrict__ xl, float* __restrict__ xr) {
    int i = blockIdx.x * blockDim.x + threadIdx.x;
    if (i >= NN) return;
    float v[IN];
    #pragma unroll
    for (int k = 0; k < IN; ++k) v[k] = in[(size_t)i * IN + k];
    #pragma unroll
    for (int j = 0; j < OUT; ++j) {
        float a = bl[j], b = br[j];
        #pragma unroll
        for (int k = 0; k < IN; ++k) { a += v[k] * Wl[k * OUT + j]; b += v[k] * Wr[k * OUT + j]; }
        xl[(size_t)i * OUT + j] = a;
        xr[(size_t)i * OUT + j] = b;
    }
}

// ---------------- GATv2 alpha helper ----------------
template <int C>
__device__ __forceinline__ void gat_alpha(int e, const int* src, const int* dst,
                                          const float* ew, const float* eamean,
                                          const float* xl, const float* xr,
                                          const float* We, const float* att,
                                          int& s, int& d, float alpha[4]) {
    constexpr int F = 4 * C;
    float eav;
    if (e < EE) { s = src[e]; d = dst[e]; eav = ew[e]; }
    else        { s = d = e - EE; eav = eamean[s]; }
    #pragma unroll
    for (int h = 0; h < 4; ++h) {
        float a = 0.f;
        #pragma unroll
        for (int c = 0; c < C; ++c) {
            int j = h * C + c;
            float m = lrelu(xl[(size_t)s * F + j] + xr[(size_t)d * F + j] + eav * We[j]);
            a += att[h * C + c] * m;
        }
        alpha[h] = a;
    }
}

template <int C>
__global__ void k_gat_max(const int* __restrict__ src, const int* __restrict__ dst,
                          const float* __restrict__ ew, const float* __restrict__ eamean,
                          const float* __restrict__ xl, const float* __restrict__ xr,
                          const float* __restrict__ We, const float* __restrict__ att,
                          unsigned* maxb) {
    int e = blockIdx.x * blockDim.x + threadIdx.x;
    if (e >= ESL) return;
    int s, d; float alpha[4];
    gat_alpha<C>(e, src, dst, ew, eamean, xl, xr, We, att, s, d, alpha);
    #pragma unroll
    for (int h = 0; h < 4; ++h) atomicMax(&maxb[(size_t)d * 4 + h], fenc(alpha[h]));
}

template <int C>
__global__ void k_gat_exp(const int* __restrict__ src, const int* __restrict__ dst,
                          const float* __restrict__ ew, const float* __restrict__ eamean,
                          const float* __restrict__ xl, const float* __restrict__ xr,
                          const float* __restrict__ We, const float* __restrict__ att,
                          const unsigned* maxb, float* __restrict__ aex, float* sumb) {
    int e = blockIdx.x * blockDim.x + threadIdx.x;
    if (e >= ESL) return;
    int s, d; float alpha[4];
    gat_alpha<C>(e, src, dst, ew, eamean, xl, xr, We, att, s, d, alpha);
    #pragma unroll
    for (int h = 0; h < 4; ++h) {
        float ex = __expf(alpha[h] - fdec(maxb[(size_t)d * 4 + h]));
        aex[(size_t)e * 4 + h] = ex;
        atomicAdd(&sumb[(size_t)d * 4 + h], ex);
    }
}

template <int C>
__global__ void k_gat_scatter(const int* __restrict__ src, const int* __restrict__ dst,
                              const float* __restrict__ xl,
                              const float* __restrict__ aex, const float* __restrict__ sumb,
                              float* outb) {
    constexpr int F = 4 * C;
    int e = blockIdx.x * blockDim.x + threadIdx.x;
    if (e >= ESL) return;
    int s, d;
    if (e < EE) { s = src[e]; d = dst[e]; } else { s = d = e - EE; }
    #pragma unroll
    for (int h = 0; h < 4; ++h) {
        float w = aex[(size_t)e * 4 + h] / (sumb[(size_t)d * 4 + h] + 1e-16f);
        #pragma unroll
        for (int c = 0; c < C; ++c) {
            int j = h * C + c;
            atomicAdd(&outb[(size_t)d * F + j], xl[(size_t)s * F + j] * w);
        }
    }
}

// ---------------- bias + relu ----------------
template <int F>
__global__ void k_bias_relu(const float* __restrict__ in, const float* __restrict__ bias,
                            float* __restrict__ out) {
    int i = blockIdx.x * blockDim.x + threadIdx.x;
    if (i >= NN * F) return;
    out[i] = relu(in[i] + bias[i % F]);
}

// ---------------- final node: relu(out2+bias2) @ Wn + bn, relu, -> f16 ----------------
__global__ void k_final_node(const float* __restrict__ out2, const float* __restrict__ bias2,
                             const float* __restrict__ Wn, const float* __restrict__ bn,
                             _Float16* __restrict__ nfh) {
    int i = blockIdx.x * blockDim.x + threadIdx.x;
    if (i >= NN) return;
    float t[32];
    #pragma unroll
    for (int j = 0; j < 32; ++j) t[j] = relu(out2[(size_t)i * 32 + j] + bias2[j]);
    #pragma unroll
    for (int j = 0; j < 32; ++j) {
        float v = bn[j];
        #pragma unroll
        for (int k = 0; k < 32; ++k) v += t[k] * Wn[k * 32 + j];
        nfh[(size_t)i * 32 + j] = (_Float16)relu(v);
    }
}

// ---------------- edge scorer: [ea | nf16[src] | nf16[dst]](65->96) @ We_fc(65x32) via WMMA ----
// 16 edges x 32 outputs per wave; K padded to 96 = 3 chunks of 32; f16 in, f32 accum.
__global__ __launch_bounds__(256) void k_scores_wmma(
        const int* __restrict__ src, const int* __restrict__ dst,
        const float* __restrict__ ew, const _Float16* __restrict__ nfh,
        const float* __restrict__ Wefc, const float* __restrict__ befc,
        float* __restrict__ out) {
    __shared__ _Float16 sfeat[8][16][96];
    const int tid = threadIdx.x;
    const int wid = tid >> 5;
    const int l   = tid & 31;
    const int g   = l >> 4;        // lane half-group (K-split per ISA 16-bit A layout)
    const int m   = l & 15;        // row (A) / col (B, D)

    // zero K pad region 64..95 once (k=64 is rewritten each tile)
    for (int idx = l; idx < 16 * 32; idx += 32)
        sfeat[wid][idx >> 5][64 + (idx & 31)] = (_Float16)0.f;

    // hoist B fragments: 3 K-chunks x 2 N-tiles, convert We_fc fp32 -> f16
    v16h Bf[3][2];
    #pragma unroll
    for (int c = 0; c < 3; ++c)
        #pragma unroll
        for (int t = 0; t < 2; ++t)
            #pragma unroll
            for (int i = 0; i < 16; ++i) {
                int kk = c * 32 + ((i & 8) << 1) + g * 8 + (i & 7);
                int nn = t * 16 + m;
                float v = (kk < 65) ? Wefc[kk * 32 + nn] : 0.f;
                Bf[c][t][i] = (_Float16)v;
            }
    const float b0 = befc[m], b1 = befc[16 + m];

    const int gw = blockIdx.x * 8 + wid;               // 0..9999 waves
    for (int tile = gw; tile < (EE / 16); tile += 10000) {
        const int e0 = tile * 16;
        const int e  = e0 + m;
        if (g == 0) {
            sfeat[wid][m][0] = (_Float16)ew[e];
            const _Float16* p = nfh + (size_t)src[e] * 32;
            #pragma unroll
            for (int j = 0; j < 32; ++j) sfeat[wid][m][1 + j] = p[j];
        } else {
            const _Float16* p = nfh + (size_t)dst[e] * 32;
            #pragma unroll
            for (int j = 0; j < 32; ++j) sfeat[wid][m][33 + j] = p[j];
        }
        asm volatile("s_wait_dscnt 0" ::: "memory");   // per-wave LDS store->load fence

        v8f acc0 = {};
        v8f acc1 = {};
        #pragma unroll
        for (int c = 0; c < 3; ++c) {
            v16h Af;
            #pragma unroll
            for (int i = 0; i < 16; ++i)
                Af[i] = sfeat[wid][m][c * 32 + ((i & 8) << 1) + g * 8 + (i & 7)];
            acc0 = __builtin_amdgcn_wmma_f32_16x16x32_f16(false, Af, false, Bf[c][0],
                                                          (short)0, acc0, false, false);
            acc1 = __builtin_amdgcn_wmma_f32_16x16x32_f16(false, Af, false, Bf[c][1],
                                                          (short)0, acc1, false, false);
        }
        // D layout: lane -> N = l&15, VGPR r -> M = r + 8*(l>>4)
        #pragma unroll
        for (int r = 0; r < 8; ++r) {
            const size_t erow = (size_t)(e0 + 8 * g + r) * 32;
            out[erow + m]      = acc0[r] + b0;
            out[erow + 16 + m] = acc1[r] + b1;
        }
    }
}

// ---------------- final segment softmax over src for columns 0,1 ----------------
__global__ void k_smax(const int* __restrict__ src, const float* __restrict__ sc, unsigned* smax) {
    int e = blockIdx.x * blockDim.x + threadIdx.x;
    if (e >= EE) return;
    int s = src[e];
    #pragma unroll
    for (int c = 0; c < 2; ++c) atomicMax(&smax[(size_t)s * 2 + c], fenc(sc[(size_t)e * 32 + c]));
}
__global__ void k_sexp(const int* __restrict__ src, const float* __restrict__ sc,
                       const unsigned* __restrict__ smax, float* __restrict__ sexp, float* ssum) {
    int e = blockIdx.x * blockDim.x + threadIdx.x;
    if (e >= EE) return;
    int s = src[e];
    #pragma unroll
    for (int c = 0; c < 2; ++c) {
        float ex = __expf(sc[(size_t)e * 32 + c] - fdec(smax[(size_t)s * 2 + c]));
        sexp[(size_t)e * 2 + c] = ex;
        atomicAdd(&ssum[(size_t)s * 2 + c], ex);
    }
}
__global__ void k_sfinal(const int* __restrict__ src, const float* __restrict__ sexp,
                         const float* __restrict__ ssum, float* __restrict__ sc) {
    int e = blockIdx.x * blockDim.x + threadIdx.x;
    if (e >= EE) return;
    int s = src[e];
    #pragma unroll
    for (int c = 0; c < 2; ++c)
        sc[(size_t)e * 32 + c] = sexp[(size_t)e * 2 + c] / (ssum[(size_t)s * 2 + c] + 1e-16f);
}

// ---------------- host ----------------
extern "C" void kernel_launch(void* const* d_in, const int* in_sizes, int n_in,
                              void* d_out, int out_size, void* d_ws, size_t ws_size,
                              hipStream_t stream) {
    const float* x      = (const float*)d_in[0];
    const int*   ei     = (const int*)  d_in[1];
    const float* ew     = (const float*)d_in[2];
    const float* W_ee   = (const float*)d_in[3];
    const float* b_ee   = (const float*)d_in[4];
    const float* W_pre  = (const float*)d_in[5];
    const float* b_pre  = (const float*)d_in[6];
    const float* W_post = (const float*)d_in[7];
    const float* b_post = (const float*)d_in[8];
    const float* W_pna  = (const float*)d_in[9];
    const float* b_pna  = (const float*)d_in[10];
    const float* W1l = (const float*)d_in[11]; const float* b1l  = (const float*)d_in[12];
    const float* W1r = (const float*)d_in[13]; const float* b1r  = (const float*)d_in[14];
    const float* W1e = (const float*)d_in[15]; const float* att1 = (const float*)d_in[16];
    const float* bias1 = (const float*)d_in[17];
    const float* W2l = (const float*)d_in[18]; const float* b2l  = (const float*)d_in[19];
    const float* W2r = (const float*)d_in[20]; const float* b2r  = (const float*)d_in[21];
    const float* W2e = (const float*)d_in[22]; const float* att2 = (const float*)d_in[23];
    const float* bias2 = (const float*)d_in[24];
    const float* Wn  = (const float*)d_in[25]; const float* bn   = (const float*)d_in[26];
    const float* Wefc = (const float*)d_in[27]; const float* befc = (const float*)d_in[28];

    const int* src = ei;
    const int* dst = ei + EE;

    char* ws = (char*)d_ws;
    size_t off = 0;
    auto alloc = [&](size_t bytes) -> void* {
        void* p = ws + off;
        off += bytes;
        off = (off + 255) & ~(size_t)255;
        return p;
    };

    float*    pna_sum = (float*)   alloc((size_t)NN * 5 * 4);
    float*    pna_sq  = (float*)   alloc((size_t)NN * 5 * 4);
    unsigned* pna_min = (unsigned*)alloc((size_t)NN * 5 * 4);
    unsigned* pna_max = (unsigned*)alloc((size_t)NN * 5 * 4);
    float*    cnt     = (float*)   alloc((size_t)NN * 4);
    float*    easum   = (float*)   alloc((size_t)NN * 4);
    float*    eamean  = (float*)   alloc((size_t)NN * 4);
    float*    nf8     = (float*)   alloc((size_t)NN * 8 * 4);
    float*    xl1     = (float*)   alloc((size_t)NN * 16 * 4);
    float*    xr1     = (float*)   alloc((size_t)NN * 16 * 4);
    unsigned* max1    = (unsigned*)alloc((size_t)NN * 4 * 4);
    float*    sum1    = (float*)   alloc((size_t)NN * 4 * 4);
    float*    out1    = (float*)   alloc((size_t)NN * 16 * 4);
    float*    nf16    = (float*)   alloc((size_t)NN * 16 * 4);
    float*    xl2     = (float*)   alloc((size_t)NN * 32 * 4);
    float*    xr2     = (float*)   alloc((size_t)NN * 32 * 4);
    unsigned* max2    = (unsigned*)alloc((size_t)NN * 4 * 4);
    float*    sum2    = (float*)   alloc((size_t)NN * 4 * 4);
    float*    out2    = (float*)   alloc((size_t)NN * 32 * 4);
    _Float16* nfh     = (_Float16*)alloc((size_t)NN * 32 * 2);
    float*    aex     = (float*)   alloc((size_t)ESL * 4 * 4);
    float*    sexp    = (float*)   alloc((size_t)EE * 2 * 4);
    unsigned* smax    = (unsigned*)alloc((size_t)NN * 2 * 4);
    float*    ssum    = (float*)   alloc((size_t)NN * 2 * 4);

    const int T = 256;
    const int gN   = (NN + T - 1) / T;
    const int gE   = (EE + T - 1) / T;
    const int gESL = (ESL + T - 1) / T;
    const int gI   = (NN * 32 + T - 1) / T;
    float* sc = (float*)d_out;

    k_init<<<gI, T, 0, stream>>>(pna_sum, pna_sq, pna_min, pna_max, cnt, easum,
                                 max1, sum1, out1, max2, sum2, out2, smax, ssum);
    // PNA
    k_pna_edge<<<gE, T, 0, stream>>>(x, src, dst, ew, W_ee, b_ee, W_pre, b_pre,
                                     pna_sum, pna_sq, pna_min, pna_max, cnt, easum);
    k_pna_node<<<gN, T, 0, stream>>>(x, pna_sum, pna_sq, pna_min, pna_max, cnt, easum,
                                     W_post, b_post, W_pna, b_pna, nf8, eamean);
    // GATv2 layer 1 (H=4, C=4)
    k_dual_linear<8, 16><<<gN, T, 0, stream>>>(nf8, W1l, b1l, W1r, b1r, xl1, xr1);
    k_gat_max<4><<<gESL, T, 0, stream>>>(src, dst, ew, eamean, xl1, xr1, W1e, att1, max1);
    k_gat_exp<4><<<gESL, T, 0, stream>>>(src, dst, ew, eamean, xl1, xr1, W1e, att1, max1, aex, sum1);
    k_gat_scatter<4><<<gESL, T, 0, stream>>>(src, dst, xl1, aex, sum1, out1);
    k_bias_relu<16><<<(NN * 16 + T - 1) / T, T, 0, stream>>>(out1, bias1, nf16);
    // GATv2 layer 2 (H=4, C=8)
    k_dual_linear<16, 32><<<gN, T, 0, stream>>>(nf16, W2l, b2l, W2r, b2r, xl2, xr2);
    k_gat_max<8><<<gESL, T, 0, stream>>>(src, dst, ew, eamean, xl2, xr2, W2e, att2, max2);
    k_gat_exp<8><<<gESL, T, 0, stream>>>(src, dst, ew, eamean, xl2, xr2, W2e, att2, max2, aex, sum2);
    k_gat_scatter<8><<<gESL, T, 0, stream>>>(src, dst, xl2, aex, sum2, out2);
    // final node transform -> f16 features
    k_final_node<<<gN, T, 0, stream>>>(out2, bias2, Wn, bn, nfh);
    // edge scorer GEMM (WMMA): 1250 blocks x 8 waves x 10 tiles = 100000 tiles of 16 edges
    k_scores_wmma<<<1250, 256, 0, stream>>>(src, dst, ew, nfh, Wefc, befc, sc);
    // segment softmax over src for score columns 0 and 1
    k_smax  <<<gE, T, 0, stream>>>(src, sc, smax);
    k_sexp  <<<gE, T, 0, stream>>>(src, sc, smax, sexp, ssum);
    k_sfinal<<<gE, T, 0, stream>>>(src, sexp, ssum, sc);
}
